// ComplexCNNTransformer_21079699488882
// MI455X (gfx1250) — compile-verified, matrix operand order checked by reading, not run on hardware
//
#include <hip/hip_runtime.h>
#include <cmath>

// ---------------- model constants ----------------
#define BATCH   64
#define SEQ     256
#define EMB     256
#define NHEADS  8
#define HDIM    32
#define FFDIM   512
#define EPS_    1e-5f
#define SCALE_  0.17677669529663687f  // 32^-0.5

// ---------------- WMMA types / helpers ----------------
typedef __attribute__((ext_vector_type(16))) __bf16 v16bf;
typedef __attribute__((ext_vector_type(8)))  float  v8f;

__device__ __forceinline__ int frag_kbase(int v, int half) {
  // ISA 7.12.2: 16-bit A 16x32 -- lanes 0-15: K 0-7 (v0-3), 16-23 (v4-7); lanes 16-31: K 8-15, 24-31
  return (v < 4) ? (half * 8 + 2 * v) : (16 + half * 8 + 2 * (v - 4));
}

// A fragment: X row-major [M,K]; rows m0..m0+15, cols k0..k0+31
__device__ __forceinline__ v16bf load_a_frag(const float* __restrict__ X, int ld, int m0, int k0) {
  int lane = threadIdx.x & 31;
  int half = lane >> 4;
  const float* row = X + (size_t)(m0 + (lane & 15)) * ld + k0;
  v16bf a;
#pragma unroll
  for (int v = 0; v < 8; ++v) {
    int kb = frag_kbase(v, half);
    a[2 * v]     = (__bf16)row[kb];
    a[2 * v + 1] = (__bf16)row[kb + 1];
  }
  return a;
}

// B fragment where B is row-major [K,N]: B[k,n] at Bm[k*ld + n]
__device__ __forceinline__ v16bf load_b_frag(const float* __restrict__ Bm, int ld, int k0, int n0) {
  int lane = threadIdx.x & 31;
  int half = lane >> 4;
  const float* col = Bm + (size_t)k0 * ld + (n0 + (lane & 15));
  v16bf b;
#pragma unroll
  for (int v = 0; v < 8; ++v) {
    int kb = frag_kbase(v, half);
    b[2 * v]     = (__bf16)col[(size_t)kb * ld];
    b[2 * v + 1] = (__bf16)col[(size_t)(kb + 1) * ld];
  }
  return b;
}

// Load real/imag fragments plus Karatsuba sum fragment (r + sgn*i) in one pass.
__device__ __forceinline__ void load_frag3(const float* __restrict__ Pr, const float* __restrict__ Pi,
                                           int ld, int row0, int k0, float sgn,
                                           v16bf& fr, v16bf& fi, v16bf& fs) {
  int lane = threadIdx.x & 31;
  int half = lane >> 4;
  const float* pr = Pr + (size_t)(row0 + (lane & 15)) * ld + k0;
  const float* pi = Pi + (size_t)(row0 + (lane & 15)) * ld + k0;
#pragma unroll
  for (int v = 0; v < 8; ++v) {
    int kb = frag_kbase(v, half);
    float r0 = pr[kb], r1 = pr[kb + 1];
    float i0 = pi[kb], i1 = pi[kb + 1];
    fr[2 * v] = (__bf16)r0;               fr[2 * v + 1] = (__bf16)r1;
    fi[2 * v] = (__bf16)i0;               fi[2 * v + 1] = (__bf16)i1;
    fs[2 * v] = (__bf16)(r0 + sgn * i0);  fs[2 * v + 1] = (__bf16)(r1 + sgn * i1);
  }
}

__device__ __forceinline__ v8f wmma_bf16(v16bf a, v16bf b, v8f c) {
  return __builtin_amdgcn_wmma_f32_16x16x32_bf16(false, a, false, b, (short)0, c, false, false);
}

#define V8Z {0.f,0.f,0.f,0.f,0.f,0.f,0.f,0.f}

// ---------------- complex GEMM: Y = X @ W^T (+bias), optional ReLU ----------------
// Karatsuba 3-product complex multiply, 32x16 output tile per wave (2x1 WMMA tiles).
// B-triple loaded once per K-step and reused by both M-subtiles; A-triples are
// transient (loaded, consumed by 3 WMMAs, released) to keep VGPR pressure < 256.
// Xr/Xi: [M,K] row-major, Wr/Wi: [N,K] row-major, Yr/Yi: [M,N].
__global__ __launch_bounds__(32)
void cgemm_wmma_kernel(const float* __restrict__ Xr, const float* __restrict__ Xi,
                       const float* __restrict__ Wr, const float* __restrict__ Wi,
                       const float* __restrict__ br, const float* __restrict__ bi,
                       float* __restrict__ Yr, float* __restrict__ Yi,
                       int M, int N, int K, int relu) {
  int m0 = blockIdx.x * 32;
  int n0 = blockIdx.y * 16;
  v8f t1[2] = {V8Z, V8Z};
  v8f t2[2] = {V8Z, V8Z};
  v8f t3[2] = {V8Z, V8Z};
  for (int k0 = 0; k0 < K; k0 += 32) {
    if (k0 + 32 < K) {  // prefetch next K-tile of the streaming activation operand
      __builtin_prefetch(Xr + (size_t)(m0 + (threadIdx.x & 31)) * K + k0 + 32, 0, 1);
      __builtin_prefetch(Xi + (size_t)(m0 + (threadIdx.x & 31)) * K + k0 + 32, 0, 1);
    }
    v16bf bwr, bwi, bws;
    load_frag3(Wr, Wi, K, n0, k0, 1.f, bwr, bwi, bws);
#pragma unroll
    for (int i = 0; i < 2; ++i) {
      v16bf axr, axi, axs;
      load_frag3(Xr, Xi, K, m0 + 16 * i, k0, 1.f, axr, axi, axs);
      t1[i] = wmma_bf16(axr, bwr, t1[i]);
      t2[i] = wmma_bf16(axi, bwi, t2[i]);
      t3[i] = wmma_bf16(axs, bws, t3[i]);
    }
  }
  int lane = threadIdx.x & 31;
  int half = lane >> 4;
  int n = n0 + (lane & 15);
  float vbr = br ? br[n] : 0.f;
  float vbi = bi ? bi[n] : 0.f;
#pragma unroll
  for (int i = 0; i < 2; ++i)
#pragma unroll
    for (int r = 0; r < 8; ++r) {
      size_t o = (size_t)(m0 + 16 * i + r + 8 * half) * N + n;
      float a = t1[i][r], b = t2[i][r], c = t3[i][r];
      float yr = a - b + vbr - vbi;          // Re = t1 - t2
      float yi = c - a - b + vbr + vbi;      // Im = t3 - t1 - t2
      if (relu) { yr = fmaxf(yr, 0.f); yi = fmaxf(yi, 0.f); }
      Yr[o] = yr; Yi[o] = yi;
    }
}

// ---------------- attention score magnitudes: smag[bh,q,k] = SCALE*|q . conj(k)| ----------------
// Conjugate Karatsuba: t1=qr.kr, t2=qi.ki, t3=(qr+qi).(kr-ki); sr=t1+t2, si=t3-t1+t2.
__global__ __launch_bounds__(32)
void attn_score_kernel(const float* __restrict__ qr, const float* __restrict__ qi,
                       const float* __restrict__ kr, const float* __restrict__ ki,
                       float* __restrict__ smag, int S, int E) {
  int bh = blockIdx.x;
  int b = bh / NHEADS, h = bh % NHEADS;
  int m0 = blockIdx.y * 16;
  int n0 = blockIdx.z * 16;
  size_t hoff = (size_t)b * S * E + (size_t)h * HDIM;
  v16bf fqr, fqi, fqs, fkr, fki, fks;
  load_frag3(qr + hoff, qi + hoff, E, m0, 0, 1.f, fqr, fqi, fqs);
  load_frag3(kr + hoff, ki + hoff, E, n0, 0, -1.f, fkr, fki, fks);
  v8f t1 = V8Z, t2 = V8Z, t3 = V8Z;
  t1 = wmma_bf16(fqr, fkr, t1);
  t2 = wmma_bf16(fqi, fki, t2);
  t3 = wmma_bf16(fqs, fks, t3);
  int lane = threadIdx.x & 31;
  int n = n0 + (lane & 15);
  int half = lane >> 4;
#pragma unroll
  for (int r = 0; r < 8; ++r) {
    float sr = t1[r] + t2[r];
    float si = t3[r] - t1[r] + t2[r];
    smag[((size_t)bh * S + m0 + r + 8 * half) * S + n] = SCALE_ * sqrtf(sr * sr + si * si);
  }
}

// ---------------- attention @ V, merged-head output in [B,S,E] ----------------
// 32 query rows x full head (32) per wave; V fragments loaded once per K-step,
// A fragments transient: 8 WMMAs per K-step, ~100 live VGPRs.
__global__ __launch_bounds__(32)
void attn_av_kernel(const float* __restrict__ aw, const float* __restrict__ vr,
                    const float* __restrict__ vi, float* __restrict__ outr,
                    float* __restrict__ outi, int S, int E) {
  int bh = blockIdx.x;
  int b = bh / NHEADS, h = bh % NHEADS;
  int m0 = blockIdx.y * 32;
  const float* A = aw + (size_t)bh * S * S;
  size_t hoff = (size_t)b * S * E + (size_t)h * HDIM;
  v8f ar[2][2] = {{V8Z, V8Z}, {V8Z, V8Z}};
  v8f ai[2][2] = {{V8Z, V8Z}, {V8Z, V8Z}};
  for (int k0 = 0; k0 < S; k0 += 32) {
    v16bf fvr[2], fvi[2];
#pragma unroll
    for (int j = 0; j < 2; ++j) {
      fvr[j] = load_b_frag(vr + hoff, E, k0, 16 * j);
      fvi[j] = load_b_frag(vi + hoff, E, k0, 16 * j);
    }
#pragma unroll
    for (int i = 0; i < 2; ++i) {
      v16bf fa = load_a_frag(A, S, m0 + 16 * i, k0);
      ar[i][0] = wmma_bf16(fa, fvr[0], ar[i][0]);
      ar[i][1] = wmma_bf16(fa, fvr[1], ar[i][1]);
      ai[i][0] = wmma_bf16(fa, fvi[0], ai[i][0]);
      ai[i][1] = wmma_bf16(fa, fvi[1], ai[i][1]);
    }
  }
  int lane = threadIdx.x & 31;
  int half = lane >> 4;
  int nl = lane & 15;
#pragma unroll
  for (int i = 0; i < 2; ++i)
#pragma unroll
    for (int j = 0; j < 2; ++j)
#pragma unroll
      for (int r = 0; r < 8; ++r) {
        size_t o = hoff + (size_t)(m0 + 16 * i + r + 8 * half) * E + 16 * j + nl;
        outr[o] = ar[i][j][r];
        outi[o] = ai[i][j][r];
      }
}

// ---------------- complex conv1d (direct) ----------------
__global__ void cconv_kernel(const float* __restrict__ xr, const float* __restrict__ xi,
                             const float* __restrict__ wr, const float* __restrict__ wi,
                             const float* __restrict__ br, const float* __restrict__ bi,
                             float* __restrict__ yr, float* __restrict__ yi,
                             int Ci, int Co, int K, int Lin, int Lout, int stride, int pad) {
  int idx = blockIdx.x * blockDim.x + threadIdx.x;
  int total = BATCH * Co * Lout;
  if (idx >= total) return;
  int l = idx % Lout;
  int co = (idx / Lout) % Co;
  int b = idx / (Lout * Co);
  float arr = 0.f, aii = 0.f, air = 0.f, ari = 0.f;
  int p0 = l * stride - pad;
  for (int ci = 0; ci < Ci; ++ci) {
    const float* xrp = xr + ((size_t)b * Ci + ci) * Lin;
    const float* xip = xi + ((size_t)b * Ci + ci) * Lin;
    const float* wrp = wr + ((size_t)co * Ci + ci) * K;
    const float* wip = wi + ((size_t)co * Ci + ci) * K;
    for (int k = 0; k < K; ++k) {
      int p = p0 + k;
      if (p < 0 || p >= Lin) continue;
      float vr = xrp[p], vi = xip[p], fr = wrp[k], fi = wip[k];
      arr += vr * fr; aii += vi * fi; air += vi * fr; ari += vr * fi;
    }
  }
  yr[idx] = arr - aii + br[co] - bi[co];
  yi[idx] = air + ari + br[co] + bi[co];
}

// ---------------- batchnorm (training-mode batch stats) ----------------
__global__ void bn_stats_kernel(const float* __restrict__ x, float* __restrict__ mean,
                                float* __restrict__ rstd, int C, int L) {
  int c = blockIdx.x;
  int t = threadIdx.x;
  float s = 0.f, ss = 0.f;
  int n = BATCH * L;
  for (int i = t; i < n; i += 256) {
    int b = i / L, l = i % L;
    float v = x[((size_t)b * C + c) * L + l];
    s += v; ss += v * v;
  }
  __shared__ float sh1[256], sh2[256];
  sh1[t] = s; sh2[t] = ss; __syncthreads();
  for (int st = 128; st > 0; st >>= 1) {
    if (t < st) { sh1[t] += sh1[t + st]; sh2[t] += sh2[t + st]; }
    __syncthreads();
  }
  if (t == 0) {
    float m = sh1[0] / n;
    float v = sh2[0] / n - m * m;
    mean[c] = m;
    rstd[c] = rsqrtf(v + EPS_);
  }
}

__global__ void bn_apply_relu_kernel(float* __restrict__ x, const float* __restrict__ mean,
                                     const float* __restrict__ rstd, const float* __restrict__ w,
                                     const float* __restrict__ b, int C, int L, int total) {
  int idx = blockIdx.x * blockDim.x + threadIdx.x;
  if (idx >= total) return;
  int c = (idx / L) % C;
  float v = (x[idx] - mean[c]) * rstd[c] * w[c] + b[c];
  x[idx] = fmaxf(v, 0.f);
}

// ---------------- SE block ----------------
__global__ void se_squeeze_kernel(const float* __restrict__ xr, const float* __restrict__ xi,
                                  float* __restrict__ sr, float* __restrict__ si, int C, int L) {
  int idx = blockIdx.x * blockDim.x + threadIdx.x;  // b*C + c
  if (idx >= BATCH * C) return;
  const float* pr = xr + (size_t)idx * L;
  const float* pi = xi + (size_t)idx * L;
  float ar = 0.f, ai = 0.f;
  for (int l = 0; l < L; ++l) { ar += pr[l]; ai += pi[l]; }
  sr[idx] = ar / L;
  si[idx] = ai / L;
}

__global__ void se_cfc_kernel(const float* __restrict__ xr, const float* __restrict__ xi,
                              const float* __restrict__ wr, const float* __restrict__ wi,
                              float* __restrict__ yr, float* __restrict__ yi,
                              int Ci, int Co, int relu) {
  int idx = blockIdx.x * blockDim.x + threadIdx.x;  // b*Co + co
  if (idx >= BATCH * Co) return;
  int b = idx / Co, co = idx % Co;
  float rr = 0.f, ii = 0.f, ir = 0.f, ri = 0.f;
  const float* xrp = xr + (size_t)b * Ci;
  const float* xip = xi + (size_t)b * Ci;
  const float* wrp = wr + (size_t)co * Ci;
  const float* wip = wi + (size_t)co * Ci;
  for (int ci = 0; ci < Ci; ++ci) {
    rr += xrp[ci] * wrp[ci]; ii += xip[ci] * wip[ci];
    ir += xip[ci] * wrp[ci]; ri += xrp[ci] * wip[ci];
  }
  float oR = rr - ii, oI = ir + ri;
  if (relu) { oR = fmaxf(oR, 0.f); oI = fmaxf(oI, 0.f); }
  yr[idx] = oR; yi[idx] = oI;
}

__global__ void se_gate_kernel(float* __restrict__ xr, float* __restrict__ xi,
                               const float* __restrict__ y2r, const float* __restrict__ y2i,
                               int C, int L) {
  int idx = blockIdx.x * blockDim.x + threadIdx.x;
  int total = BATCH * C * L;
  if (idx >= total) return;
  int bc = idx / L;
  float pr = y2r[bc], pi = y2i[bc];
  float mag = sqrtf(pr * pr + pi * pi + 1e-8f);
  float ph = atan2f(pi, pr);
  float wgt = 1.f / (1.f + expf(-mag));
  float gr = wgt * cosf(ph), gi = wgt * sinf(ph);
  float a = xr[idx], b = xi[idx];
  xr[idx] = a * gr - b * gi;
  xi[idx] = a * gi + b * gr;
}

// ---------------- transpose [B,C,S] -> [B,S,C] + positional ----------------
__global__ void transpose_pos_kernel(const float* __restrict__ xr, const float* __restrict__ xi,
                                     const float* __restrict__ posr, const float* __restrict__ posi,
                                     float* __restrict__ yr, float* __restrict__ yi) {
  int idx = blockIdx.x * blockDim.x + threadIdx.x;  // b*S*E + s*E + e
  int total = BATCH * SEQ * EMB;
  if (idx >= total) return;
  int e = idx % EMB;
  int s = (idx / EMB) % SEQ;
  int b = idx / (EMB * SEQ);
  size_t src = ((size_t)b * EMB + e) * SEQ + s;  // [B,C=E,S]
  yr[idx] = xr[src] + posr[s * EMB + e];
  yi[idx] = xi[src] + posi[s * EMB + e];
}

// ---------------- shared reduction helper (256 threads) ----------------
__device__ __forceinline__ float block_sum_256(float v, float* sh) {
  int t = threadIdx.x;
  sh[t] = v; __syncthreads();
  for (int s = 128; s > 0; s >>= 1) {
    if (t < s) sh[t] += sh[t + s];
    __syncthreads();
  }
  float r = sh[0]; __syncthreads();
  return r;
}

// ---------------- complex LayerNorm (over E=256), optional residual ----------------
__global__ void cln_kernel(const float* __restrict__ xr, const float* __restrict__ xi,
                           const float* __restrict__ resr, const float* __restrict__ resi,
                           const float* __restrict__ wr, const float* __restrict__ br,
                           const float* __restrict__ wi, const float* __restrict__ bi,
                           float* __restrict__ yr, float* __restrict__ yi) {
  int row = blockIdx.x;
  int e = threadIdx.x;
  size_t o = (size_t)row * EMB + e;
  float vr = xr[o] + (resr ? resr[o] : 0.f);
  float vi = xi[o] + (resi ? resi[o] : 0.f);
  __shared__ float sh[256];
  float sr = block_sum_256(vr, sh);
  float srr = block_sum_256(vr * vr, sh);
  float si = block_sum_256(vi, sh);
  float sii = block_sum_256(vi * vi, sh);
  float mr = sr / EMB, mi = si / EMB;
  float rstd_r = rsqrtf(srr / EMB - mr * mr + EPS_);
  float rstd_i = rsqrtf(sii / EMB - mi * mi + EPS_);
  yr[o] = (vr - mr) * rstd_r * wr[e] + br[e];
  yi[o] = (vi - mi) * rstd_i * wi[e] + bi[e];
}

// ---------------- row softmax over S=256 ----------------
__global__ void softmax_kernel(float* __restrict__ s) {
  int row = blockIdx.x;
  int t = threadIdx.x;
  size_t o = (size_t)row * SEQ + t;
  float v = s[o];
  __shared__ float sh[256];
  sh[t] = v; __syncthreads();
  for (int st = 128; st > 0; st >>= 1) {
    if (t < st) sh[t] = fmaxf(sh[t], sh[t + st]);
    __syncthreads();
  }
  float mx = sh[0]; __syncthreads();
  float e = expf(v - mx);
  float sum = block_sum_256(e, sh);
  s[o] = e / sum;
}

// ---------------- mean-pool + magnitude + fc + sigmoid ----------------
__global__ void pool_fc_kernel(const float* __restrict__ xr, const float* __restrict__ xi,
                               const float* __restrict__ fw, const float* __restrict__ fb,
                               float* __restrict__ out) {
  int b = blockIdx.x;
  int e = threadIdx.x;
  float sr = 0.f, si = 0.f;
  for (int s = 0; s < SEQ; ++s) {
    size_t o = ((size_t)b * SEQ + s) * EMB + e;
    sr += xr[o]; si += xi[o];
  }
  sr /= SEQ; si /= SEQ;
  float mag = sqrtf(sr * sr + si * si);
  __shared__ float sh[256];
  float dot = block_sum_256(mag * fw[e], sh);
  if (e == 0) out[b] = 1.f / (1.f + expf(-(dot + fb[0])));
}

// ================= host launcher =================
extern "C" void kernel_launch(void* const* d_in, const int* in_sizes, int n_in,
                              void* d_out, int out_size, void* d_ws, size_t ws_size,
                              hipStream_t stream) {
  (void)in_sizes; (void)n_in; (void)out_size; (void)ws_size;
#define PF(i) ((const float*)d_in[(i)])
  const float* xr_in = PF(0);
  const float* xi_in = PF(1);

  // workspace bump allocator
  float* W = (float*)d_ws;
  size_t off = 0;
  auto alloc = [&](size_t n) { float* p = W + off; off += n; return p; };
  const size_t ACT = (size_t)BATCH * 256 * 256;  // 4,194,304 (every conv output & [B,S,E] act)
  float* bufAr = alloc(ACT); float* bufAi = alloc(ACT);
  float* bufBr = alloc(ACT); float* bufBi = alloc(ACT);
  float* meanR = alloc(256); float* rstdR = alloc(256);
  float* meanI = alloc(256); float* rstdI = alloc(256);
  float* sqr = alloc(BATCH * 256); float* sqi = alloc(BATCH * 256);
  float* y1r = alloc(BATCH * 16);  float* y1i = alloc(BATCH * 16);
  float* y2r = alloc(BATCH * 256); float* y2i = alloc(BATCH * 256);
  float* qr = alloc(ACT); float* qi = alloc(ACT);
  float* kr = alloc(ACT); float* ki = alloc(ACT);
  float* vr = alloc(ACT); float* vi = alloc(ACT);
  float* tor_ = alloc(ACT); float* toi = alloc(ACT);
  float* hr = alloc((size_t)BATCH * SEQ * FFDIM); float* hi = alloc((size_t)BATCH * SEQ * FFDIM);
  float* scores = alloc((size_t)BATCH * NHEADS * SEQ * SEQ);

  // ---- conv stack (param leaf order: pytree sorted keys; each sub-dict {bi,br,wi,wr}) ----
  struct CC { int ci, co, k, st, pad, lin, lout, wb, bnb; };
  const CC convs[5] = {
    {2,   16, 7, 2, 3, 8192, 4096, 22, 2},
    {16,  32, 5, 2, 2, 4096, 2048, 26, 6},
    {32,  64, 3, 2, 1, 2048, 1024, 30, 10},
    {64, 128, 3, 2, 1, 1024, 512,  34, 14},
    {128,256, 1, 2, 0, 512,  256,  38, 18},
  };
  const float* cr = xr_in; const float* ci = xi_in;
  for (int i = 0; i < 5; ++i) {
    const CC& c = convs[i];
    float* dr = (i % 2 == 0) ? bufAr : bufBr;
    float* di = (i % 2 == 0) ? bufAi : bufBi;
    int total = BATCH * c.co * c.lout;
    int blk = (total + 255) / 256;
    cconv_kernel<<<blk, 256, 0, stream>>>(cr, ci, PF(c.wb + 3), PF(c.wb + 2), PF(c.wb + 1),
                                          PF(c.wb + 0), dr, di, c.ci, c.co, c.k, c.lin, c.lout,
                                          c.st, c.pad);
    bn_stats_kernel<<<c.co, 256, 0, stream>>>(dr, meanR, rstdR, c.co, c.lout);
    bn_stats_kernel<<<c.co, 256, 0, stream>>>(di, meanI, rstdI, c.co, c.lout);
    bn_apply_relu_kernel<<<blk, 256, 0, stream>>>(dr, meanR, rstdR, PF(c.bnb + 3), PF(c.bnb + 1),
                                                  c.co, c.lout, total);
    bn_apply_relu_kernel<<<blk, 256, 0, stream>>>(di, meanI, rstdI, PF(c.bnb + 2), PF(c.bnb + 0),
                                                  c.co, c.lout, total);
    cr = dr; ci = di;
  }
  // after conv5 activations live in bufA: [B, 256, 256]

  // ---- SE block (fc1 wi=114,wr=115; fc2 wi=116,wr=117) ----
  se_squeeze_kernel<<<(BATCH * 256) / 256, 256, 0, stream>>>(bufAr, bufAi, sqr, sqi, 256, 256);
  se_cfc_kernel<<<(BATCH * 16 + 255) / 256, 256, 0, stream>>>(sqr, sqi, PF(115), PF(114),
                                                              y1r, y1i, 256, 16, 1);
  se_cfc_kernel<<<(BATCH * 256) / 256, 256, 0, stream>>>(y1r, y1i, PF(117), PF(116),
                                                         y2r, y2i, 16, 256, 0);
  se_gate_kernel<<<(int)(ACT / 256), 256, 0, stream>>>(bufAr, bufAi, y2r, y2i, 256, 256);

  // ---- transpose + positional (pos_i=112, pos_r=113), then LN (norm: bi=108,br=109,wi=110,wr=111)
  transpose_pos_kernel<<<(int)(ACT / 256), 256, 0, stream>>>(bufAr, bufAi, PF(113), PF(112),
                                                             bufBr, bufBi);
  cln_kernel<<<BATCH * SEQ, 256, 0, stream>>>(bufBr, bufBi, nullptr, nullptr,
                                              PF(111), PF(109), PF(110), PF(108), bufAr, bufAi);

  // ---- transformer layers ----
  const int M = BATCH * SEQ;  // 16384
  dim3 gP(M / 32, EMB / 16);          // 32x16 tiles: projections 256x256
  dim3 gF1(M / 32, FFDIM / 16);       // 256 -> 512
  dim3 gS(BATCH * NHEADS, SEQ / 16, SEQ / 16);
  dim3 gAV(BATCH * NHEADS, SEQ / 32);
  for (int lyr = 0; lyr < 2; ++lyr) {
    int base = 44 + 32 * lyr;
    // in-layer offsets (sorted keys): k:0 l1:4 l2:8 n1:12 n2:16 o:20 q:24 v:28; sub {bi,br,wi,wr}
    const float *k_bi = PF(base+0),  *k_br = PF(base+1),  *k_wi = PF(base+2),  *k_wr = PF(base+3);
    const float *l1bi = PF(base+4),  *l1br = PF(base+5),  *l1wi = PF(base+6),  *l1wr = PF(base+7);
    const float *l2bi = PF(base+8),  *l2br = PF(base+9),  *l2wi = PF(base+10), *l2wr = PF(base+11);
    const float *n1bi = PF(base+12), *n1br = PF(base+13), *n1wi = PF(base+14), *n1wr = PF(base+15);
    const float *n2bi = PF(base+16), *n2br = PF(base+17), *n2wi = PF(base+18), *n2wr = PF(base+19);
    const float *o_bi = PF(base+20), *o_br = PF(base+21), *o_wi = PF(base+22), *o_wr = PF(base+23);
    const float *q_bi = PF(base+24), *q_br = PF(base+25), *q_wi = PF(base+26), *q_wr = PF(base+27);
    const float *v_bi = PF(base+28), *v_br = PF(base+29), *v_wi = PF(base+30), *v_wr = PF(base+31);

    // Q/K/V projections (WMMA, Karatsuba)
    cgemm_wmma_kernel<<<gP, 32, 0, stream>>>(bufAr, bufAi, q_wr, q_wi, q_br, q_bi, qr, qi, M, EMB, EMB, 0);
    cgemm_wmma_kernel<<<gP, 32, 0, stream>>>(bufAr, bufAi, k_wr, k_wi, k_br, k_bi, kr, ki, M, EMB, EMB, 0);
    cgemm_wmma_kernel<<<gP, 32, 0, stream>>>(bufAr, bufAi, v_wr, v_wi, v_br, v_bi, vr, vi, M, EMB, EMB, 0);
    // attention
    attn_score_kernel<<<gS, 32, 0, stream>>>(qr, qi, kr, ki, scores, SEQ, EMB);
    softmax_kernel<<<BATCH * NHEADS * SEQ, 256, 0, stream>>>(scores);
    attn_av_kernel<<<gAV, 32, 0, stream>>>(scores, vr, vi, tor_, toi, SEQ, EMB);
    // output projection (reuse qr/qi as attn-out)
    cgemm_wmma_kernel<<<gP, 32, 0, stream>>>(tor_, toi, o_wr, o_wi, o_br, o_bi, qr, qi, M, EMB, EMB, 0);
    // n1: LN(x + attn) -> bufB
    cln_kernel<<<M, 256, 0, stream>>>(bufAr, bufAi, qr, qi, n1wr, n1br, n1wi, n1bi, bufBr, bufBi);
    // FFN
    cgemm_wmma_kernel<<<gF1, 32, 0, stream>>>(bufBr, bufBi, l1wr, l1wi, l1br, l1bi, hr, hi, M, FFDIM, EMB, 1);
    cgemm_wmma_kernel<<<gP, 32, 0, stream>>>(hr, hi, l2wr, l2wi, l2br, l2bi, tor_, toi, M, EMB, FFDIM, 0);
    // n2: LN(bufB + ffn) -> bufA (next layer input)
    cln_kernel<<<M, 256, 0, stream>>>(bufBr, bufBi, tor_, toi, n2wr, n2br, n2wi, n2bi, bufAr, bufAi);
  }

  // ---- pool + magnitude + fc (fc: b=42, w=43) + sigmoid ----
  pool_fc_kernel<<<BATCH, 256, 0, stream>>>(bufAr, bufAi, PF(43), PF(42), (float*)d_out);
#undef PF
}
